// BinaryDense_360777253278
// MI455X (gfx1250) — compile-verified, matrix operand order checked by reading, not run on hardware
//
#include <hip/hip_runtime.h>
#include <hip/hip_bf16.h>

typedef __attribute__((ext_vector_type(8)))  __bf16 v8bf;
typedef __attribute__((ext_vector_type(16))) __bf16 v16bf;
typedef __attribute__((ext_vector_type(8)))  float  v8f;

#define BDIM 4096
#define IDIM 4096
#define ODIM 4096
#define KMAT 4

union FragU { v16bf v; v8bf h[2]; };

static __device__ __forceinline__ __bf16 f2bf(float f) {
    // round-to-nearest-even f32 -> bf16 via bit ops
    unsigned u = __float_as_uint(f);
    unsigned r = (u + 0x7FFFu + ((u >> 16) & 1u)) >> 16;
    return __builtin_bit_cast(__bf16, (unsigned short)r);
}

// LDS byte offset from a generic pointer to a __shared__ object.
// ISA: the low 32 bits of a flat LDS-aperture address are the LDS offset.
static __device__ __forceinline__ unsigned lds_off(const void* p) {
    return (unsigned)(unsigned long long)p;
}

// CDNA5 async cache->LDS copy, 16B per lane, tracked by ASYNCcnt.
static __device__ __forceinline__ void async_copy_b128(unsigned lds_addr, const void* gptr) {
    asm volatile("global_load_async_to_lds_b128 %0, %1, off"
                 :: "v"(lds_addr), "v"(gptr)
                 : "memory");
}

static __device__ __forceinline__ void wait_asynccnt0() {
    asm volatile("s_wait_asynccnt 0x0" ::: "memory");
}

// ---------------- Phase 1a: x (f32) -> bf16, elementwise, vector x8 ----------------
__global__ __launch_bounds__(256) void bd_cvt_x_kernel(const float* __restrict__ x,
                                                       __bf16* __restrict__ xbf) {
    size_t idx = ((size_t)blockIdx.x * 256 + threadIdx.x) * 8;
    float4 lo = *(const float4*)(x + idx);
    float4 hi = *(const float4*)(x + idx + 4);
    v8bf o;
    o[0] = f2bf(lo.x); o[1] = f2bf(lo.y); o[2] = f2bf(lo.z); o[3] = f2bf(lo.w);
    o[4] = f2bf(hi.x); o[5] = f2bf(hi.y); o[6] = f2bf(hi.z); o[7] = f2bf(hi.w);
    *(v8bf*)(xbf + idx) = o;
}

// ---------------- Phase 1b: fold + transpose ----------------
// Wt[o][i] = bf16( sum_k sign(b[k][i][o]) * a[k][o] )
__global__ __launch_bounds__(256) void bd_fold_kernel(const float* __restrict__ b,
                                                      const float* __restrict__ a,
                                                      __bf16* __restrict__ wT) {
    __shared__ float tile[32][33];   // [o_local][i_local], padded vs bank conflicts
    const int i0 = blockIdx.x * 32;
    const int o0 = blockIdx.y * 32;
    const int tx = threadIdx.x & 31;
    const int ty = threadIdx.x >> 5;
    const int o  = o0 + tx;

    float av[KMAT];
    #pragma unroll
    for (int k = 0; k < KMAT; ++k) av[k] = a[(size_t)k * ODIM + o];

    #pragma unroll
    for (int rr = 0; rr < 4; ++rr) {
        const int il = ty + rr * 8;
        const int i  = i0 + il;
        float acc = 0.f;
        #pragma unroll
        for (int k = 0; k < KMAT; ++k) {
            float v = b[((size_t)k * IDIM + i) * ODIM + o];
            float s = (v > 0.f) ? 1.f : ((v < 0.f) ? -1.f : 0.f);
            acc += s * av[k];
        }
        tile[tx][il] = acc;
    }
    __syncthreads();
    #pragma unroll
    for (int rr = 0; rr < 4; ++rr) {
        const int ol = ty + rr * 8;
        wT[(size_t)(o0 + ol) * IDIM + (i0 + tx)] = f2bf(tile[ol][tx]);
    }
}

// ---------------- Phase 2: bf16 WMMA GEMM, f32 accumulate ----------------
// out[m][n] = sum_i X[m][i] * Wt[n][i] + bias[n]
// Block tile 128x128, BK=32, 256 threads = 8 waves (4x2), wave tile 32x64 = 2x4 WMMA tiles.
// Double-buffered LDS filled by GLOBAL_LOAD_ASYNC_TO_LDS_B128 (ASYNCcnt path).
__global__ __launch_bounds__(256) void bd_gemm_kernel(const __bf16* __restrict__ X,
                                                      const __bf16* __restrict__ Wt,
                                                      const float* __restrict__ bias,
                                                      float* __restrict__ out) {
    constexpr int BM = 128, BN = 128, BK = 32;
    __shared__ alignas(16) __bf16 Xs[2][BM][BK];   // 2 x 8 KB
    __shared__ alignas(16) __bf16 Ws[2][BN][BK];   // 2 x 8 KB

    const int m0 = blockIdx.y * BM;
    const int n0 = blockIdx.x * BN;
    const int tid  = threadIdx.x;
    const int wave = tid >> 5;
    const int lane = tid & 31;
    const int wm = wave & 3;        // M offset wm*32
    const int wn = wave >> 2;       // N offset wn*64
    const int half = lane >> 4;     // K split per ISA 16-bit A/B layout
    const int lr   = lane & 15;
    const int koff = half * 8;      // lanes 0-15: K=0..7 & 16..23 ; lanes 16-31: K=8..15 & 24..31

    const int ldr = tid >> 2;           // 0..63 staging row
    const int ldc = (tid & 3) * 8;      // K segment (8 bf16 = 16 B)

    // per-thread global sources (row fixed, k advances)
    const __bf16* gX0 = &X [(size_t)(m0 + ldr     ) * IDIM + ldc];
    const __bf16* gX1 = &X [(size_t)(m0 + ldr + 64) * IDIM + ldc];
    const __bf16* gW0 = &Wt[(size_t)(n0 + ldr     ) * IDIM + ldc];
    const __bf16* gW1 = &Wt[(size_t)(n0 + ldr + 64) * IDIM + ldc];

    // per-thread LDS destinations for both buffers
    unsigned lx0[2], lx1[2], lw0[2], lw1[2];
    #pragma unroll
    for (int bufi = 0; bufi < 2; ++bufi) {
        lx0[bufi] = lds_off(&Xs[bufi][ldr     ][ldc]);
        lx1[bufi] = lds_off(&Xs[bufi][ldr + 64][ldc]);
        lw0[bufi] = lds_off(&Ws[bufi][ldr     ][ldc]);
        lw1[bufi] = lds_off(&Ws[bufi][ldr + 64][ldc]);
    }

    v8f acc[2][4] = {};

    // ---- prologue: stage tile 0 into buffer 0
    async_copy_b128(lx0[0], gX0);
    async_copy_b128(lx1[0], gX1);
    async_copy_b128(lw0[0], gW0);
    async_copy_b128(lw1[0], gW1);
    wait_asynccnt0();
    __syncthreads();

    int buf = 0;
    for (int k0 = 0; k0 < IDIM; k0 += BK) {
        // ---- issue async DMA for next tile into the other buffer
        // (last iteration wraps to k=0: harmless dummy prefetch, keeps EXEC uniform)
        const int knext = (k0 + BK) & (IDIM - 1);
        const int nb = buf ^ 1;
        async_copy_b128(lx0[nb], gX0 + knext);
        async_copy_b128(lx1[nb], gX1 + knext);
        async_copy_b128(lw0[nb], gW0 + knext);
        async_copy_b128(lw1[nb], gW1 + knext);

        // ---- fragments from current buffer (contiguous 16B chunks per ISA VGPR layout)
        FragU af[2], bfr[4];
        #pragma unroll
        for (int tm = 0; tm < 2; ++tm) {
            const int row = wm * 32 + tm * 16 + lr;
            af[tm].h[0] = *(const v8bf*)&Xs[buf][row][koff];
            af[tm].h[1] = *(const v8bf*)&Xs[buf][row][koff + 16];
        }
        #pragma unroll
        for (int tn = 0; tn < 4; ++tn) {
            const int col = wn * 64 + tn * 16 + lr;
            bfr[tn].h[0] = *(const v8bf*)&Ws[buf][col][koff];
            bfr[tn].h[1] = *(const v8bf*)&Ws[buf][col][koff + 16];
        }

        // ---- 8 WMMAs per wave per K-step (runs in the shadow of the async DMA)
        #pragma unroll
        for (int tm = 0; tm < 2; ++tm)
            #pragma unroll
            for (int tn = 0; tn < 4; ++tn)
                acc[tm][tn] = __builtin_amdgcn_wmma_f32_16x16x32_bf16(
                    false, af[tm].v, false, bfr[tn].v,
                    (short)0, acc[tm][tn], false, false);

        // ---- drain this wave's async copies, then make them visible block-wide
        wait_asynccnt0();
        __syncthreads();
        buf = nb;
    }

    // ---- epilogue: C layout = VGPR r -> M = r + 8*half ; lane lr -> N
    #pragma unroll
    for (int tm = 0; tm < 2; ++tm) {
        const int mbase = m0 + wm * 32 + tm * 16 + half * 8;
        #pragma unroll
        for (int tn = 0; tn < 4; ++tn) {
            const int nidx = n0 + wn * 64 + tn * 16 + lr;
            const float bv = bias[nidx];
            #pragma unroll
            for (int r = 0; r < 8; ++r)
                out[(size_t)(mbase + r) * ODIM + nidx] = acc[tm][tn][r] + bv;
        }
    }
}

extern "C" void kernel_launch(void* const* d_in, const int* in_sizes, int n_in,
                              void* d_out, int out_size, void* d_ws, size_t ws_size,
                              hipStream_t stream) {
    (void)in_sizes; (void)n_in; (void)out_size; (void)ws_size;
    const float* x    = (const float*)d_in[0];   // [B, I]
    const float* b    = (const float*)d_in[1];   // [K, I, O]
    const float* a    = (const float*)d_in[2];   // [K, O]
    const float* bias = (const float*)d_in[3];   // [O]

    __bf16* xbf = (__bf16*)d_ws;                 // 32 MB
    __bf16* wT  = xbf + (size_t)BDIM * IDIM;     // 32 MB, transposed effective weight

    bd_cvt_x_kernel<<<(size_t)BDIM * IDIM / (256 * 8), 256, 0, stream>>>(x, xbf);
    bd_fold_kernel<<<dim3(IDIM / 32, ODIM / 32), 256, 0, stream>>>(b, a, wT);
    bd_gemm_kernel<<<dim3(ODIM / 128, BDIM / 128), 256, 0, stream>>>(xbf, wT, bias, (float*)d_out);
}
